// FCOS_8289286881350
// MI455X (gfx1250) — matrix operand order repaired and checked by reading, not compiled, not run on hardware
//
#include <hip/hip_runtime.h>
#include <hip/hip_bf16.h>

typedef _Float16 half8 __attribute__((ext_vector_type(8)));
typedef _Float16 v16h  __attribute__((ext_vector_type(16)));
typedef float    v8f   __attribute__((ext_vector_type(8)));
typedef unsigned int u32x4 __attribute__((ext_vector_type(4)));
typedef int          i32x4 __attribute__((ext_vector_type(4)));
typedef int          i32x8 __attribute__((ext_vector_type(8)));

// ---------------------------------------------------------------------------
// TDM helper: DMA one 32(K-rows) x 64(N-cols) f16 weight sub-panel from
// global memory into LDS. 2D descriptor: data_size=2B, tile_dim0=64,
// tile_dim1=32, tensor_dim0_stride=256 elements. LDS padding enabled:
// 4 DWORDs (16B) inserted every 32 DWORDs (128B) -> 144B LDS row stride,
// keeping each padded LDS row 16B-aligned.
// ---------------------------------------------------------------------------
__device__ __forceinline__ void tdm_load_b_chunk(unsigned lds_addr,
                                                 unsigned long long gaddr)
{
    const unsigned D0 = 1u << 20;   // huge tensor dims: no OOB clamping
    const unsigned D1 = 1u << 20;
    u32x4 g0;
    g0[0] = 1u;                                            // count=1, no gather
    g0[1] = lds_addr;                                      // LDS byte address
    g0[2] = (unsigned)(gaddr & 0xFFFFFFFFu);               // global addr lo
    g0[3] = (unsigned)((gaddr >> 32) & 0x01FFFFFFu)        // global addr hi
            | (2u << 30);                                  // type = image
    i32x8 g1;
    g1[0] = (int)((1u << 16)      // data_size = 2 bytes
                | (1u << 20)      // pad_enable
                | (4u << 22)      // pad_interval = 32 DWORDs (128B)
                | (3u << 25));    // pad_amount   = 4 DWORDs (16B)
    g1[1] = (int)((D0 & 0xFFFFu) << 16);                   // tensor_dim0 lo
    g1[2] = (int)(((D0 >> 16) & 0xFFFFu) | ((D1 & 0xFFFFu) << 16));
    g1[3] = (int)(((D1 >> 16) & 0xFFFFu) | (64u << 16));   // tile_dim0 = 64
    g1[4] = 32;                                            // tile_dim1 = 32
    g1[5] = 256;                                           // dim0 stride (elts)
    g1[6] = 0;
    g1[7] = 0;
    i32x4 z4 = {};
#if __clang_major__ >= 23
    i32x8 z8 = {};
    __builtin_amdgcn_tensor_load_to_lds(g0, g1, z4, z4, z8, 0);
#else
    __builtin_amdgcn_tensor_load_to_lds(g0, g1, z4, z4, 0);
#endif
}

// ---------------------------------------------------------------------------
// Tower conv (Cout=256): implicit-GEMM 3x3 via v_wmma_f32_16x16x32_f16 with
// the weight (B) panel streamed through wave-private double-buffered LDS by
// the Tensor Data Mover. One wave: 16 pixels x 64 out-channels (4 N-tiles).
// K-chunks of 32 march linearly over k = tap*256 + ci (72 chunks).
// Pipeline: issue TDM chunk i+1 -> wait tensorcnt<=1 (chunk i resident) ->
// ds_load B fragments + 4 WMMAs -> s_wait_dscnt 0 -> buffer recyclable.
// ---------------------------------------------------------------------------
#define TDM_ROWB 144              // padded LDS row stride in bytes
#define TDM_BUFB (32 * TDM_ROWB)  // one chunk buffer: 4608B

__global__ void __launch_bounds__(128) conv3x3_wmma_tdm(
    const _Float16* __restrict__ X, const _Float16* __restrict__ Wp,
    const _Float16* __restrict__ Zrow,
    const float* __restrict__ bias, float* __restrict__ Y,
    int HW, int W)
{
    __shared__ __align__(16) unsigned char s_b[4][2][TDM_BUFB];

    const int H      = HW / W;
    const int lane   = threadIdx.x & 31;
    const int wav    = threadIdx.x >> 5;
    const int mtiles = (HW + 15) >> 4;
    const int gw     = blockIdx.x * (blockDim.x >> 5) + wav;
    if (gw >= mtiles * 4) return;                 // wave-uniform exit

    const int ng    = gw & 3;                     // 4 N-groups of 64 channels
    const int mt    = gw >> 2;
    const int n0    = ng << 6;
    const int pbase = mt << 4;

    const int  m   = lane & 15;
    const int  klo = (lane >> 4) << 3;            // 0 or 8
    const int  p   = pbase + m;
    const bool pin = p < HW;
    const int  h   = p / W;
    const int  w   = p % W;

    // Per-wave TDM state: global base of this wave's 64-wide weight panel.
    const unsigned long long gbase =
        (unsigned long long)(unsigned long long)(const void*)(Wp + n0);
    unsigned ldsa[2];
    ldsa[0] = (unsigned)(unsigned long long)(const void*)&s_b[wav][0][0];
    ldsa[1] = (unsigned)(unsigned long long)(const void*)&s_b[wav][1][0];

    v8f zero = {};
    v8f acc[4];
    #pragma unroll
    for (int t = 0; t < 4; ++t) acc[t] = zero;

    // Prologue: start chunk 0 (k-rows 0..31) into buffer 0.
    tdm_load_b_chunk(ldsa[0], gbase);

    #pragma unroll
    for (int tap = 0; tap < 9; ++tap) {
        const int  dy    = tap / 3 - 1;
        const int  dx    = tap % 3 - 1;
        const int  hh    = h + dy;
        const int  ww    = w + dx;
        const bool valid = pin && ((unsigned)hh < (unsigned)H) &&
                                  ((unsigned)ww < (unsigned)W);
        const _Float16* xrow = valid ? (X + (size_t)(hh * W + ww) * 256 + klo)
                                     : (Zrow + klo);

        #pragma unroll
        for (int csi = 0; csi < 8; ++csi) {
            const int i   = tap * 8 + csi;        // linear K-chunk index
            const int buf = i & 1;
            if (i + 1 < 72) {                     // stream next chunk
                tdm_load_b_chunk(ldsa[buf ^ 1],
                                 gbase + (unsigned long long)(i + 1) * 32ull * 512ull);
                __builtin_amdgcn_s_wait_tensorcnt(1);   // chunk i resident
            } else {
                __builtin_amdgcn_s_wait_tensorcnt(0);   // last chunk resident
            }

            // A fragment: two 16B loads from the activation row.
            const int cs = csi * 32;
            half8 lo = *(const half8*)(xrow + cs);
            half8 hi = *(const half8*)(xrow + cs + 16);
            v16h a;
            #pragma unroll
            for (int e = 0; e < 8; ++e) { a[e] = lo[e]; a[8 + e] = hi[e]; }

            // B fragments from LDS: lane holds column N = lane&15, K-rows klo+e and 16+klo+e.
            const unsigned char* brow = &s_b[wav][buf][klo * TDM_ROWB + (lane & 15) * 2];
            #pragma unroll
            for (int t = 0; t < 4; ++t) {
                v16h b;
                #pragma unroll
                for (int e = 0; e < 8; ++e) {
                    b[e]     = *(const _Float16*)(brow + e * TDM_ROWB + t * 32);
                    b[8 + e] = *(const _Float16*)(brow + (16 + e) * TDM_ROWB + t * 32);
                }
                acc[t] = __builtin_amdgcn_wmma_f32_16x16x32_f16(
                             false, a, false, b, (short)0, acc[t], false, false);
            }
            // Fragment reads must finish before this buffer is re-filled.
            asm volatile("s_wait_dscnt 0" ::: "memory");
        }
    }

    // D layout: VGPR r, lane L -> M = r + 8*(L>=16), N = L&15
    const int mhi = (lane >> 4) << 3;
    #pragma unroll
    for (int t = 0; t < 4; ++t) {
        const int   n  = n0 + t * 16 + (lane & 15);
        const float bv = bias[n];
        #pragma unroll
        for (int r = 0; r < 8; ++r) {
            const int pp = pbase + r + mhi;
            if (pp < HW) Y[(size_t)pp * 256 + n] = acc[t][r] + bv;
        }
    }
}

// ---------------------------------------------------------------------------
// Register-direct conv (head convs, Cout=80/16): same GEMM mapping, B loaded
// straight from global (panels are tiny and L2-resident).
// ---------------------------------------------------------------------------
template <int NT>
__global__ void __launch_bounds__(128) conv3x3_wmma(
    const _Float16* __restrict__ X, const _Float16* __restrict__ Wp,
    const _Float16* __restrict__ Zrow,
    const float* __restrict__ bias, float* __restrict__ Y,
    int HW, int W, int Cout, int ldc, int CoutValid)
{
    const int H       = HW / W;
    const int lane    = threadIdx.x & 31;
    const int wav     = threadIdx.x >> 5;
    const int ngroups = Cout / (16 * NT);
    const int mtiles  = (HW + 15) >> 4;
    const int gw      = blockIdx.x * (blockDim.x >> 5) + wav;
    if (gw >= mtiles * ngroups) return;

    const int ng    = gw % ngroups;
    const int mt    = gw / ngroups;
    const int n0    = ng * 16 * NT;
    const int pbase = mt << 4;

    const int  m   = lane & 15;
    const int  klo = (lane >> 4) << 3;
    const int  p   = pbase + m;
    const bool pin = p < HW;
    const int  h   = p / W;
    const int  w   = p % W;

    v8f zero = {};
    v8f acc[NT];
    #pragma unroll
    for (int t = 0; t < NT; ++t) acc[t] = zero;

    #pragma unroll
    for (int tap = 0; tap < 9; ++tap) {
        const int  dy    = tap / 3 - 1;
        const int  dx    = tap % 3 - 1;
        const int  hh    = h + dy;
        const int  ww    = w + dx;
        const bool valid = pin && ((unsigned)hh < (unsigned)H) &&
                                  ((unsigned)ww < (unsigned)W);
        const _Float16* xrow = valid ? (X + (size_t)(hh * W + ww) * 256 + klo)
                                     : (Zrow + klo);
        const _Float16* wrow = Wp + ((size_t)tap * 256 + klo) * Cout + n0 + (lane & 15);
        if (tap < 8)
            __builtin_prefetch(Wp + ((size_t)(tap + 1) * 256 + lane) * Cout + n0, 0, 1);

        #pragma unroll
        for (int cs = 0; cs < 256; cs += 32) {
            half8 lo = *(const half8*)(xrow + cs);
            half8 hi = *(const half8*)(xrow + cs + 16);
            v16h a;
            #pragma unroll
            for (int e = 0; e < 8; ++e) { a[e] = lo[e]; a[8 + e] = hi[e]; }
            #pragma unroll
            for (int t = 0; t < NT; ++t) {
                v16h b;
                #pragma unroll
                for (int e = 0; e < 8; ++e) {
                    b[e]     = wrow[(size_t)(cs + e) * Cout + t * 16];
                    b[8 + e] = wrow[(size_t)(cs + 16 + e) * Cout + t * 16];
                }
                acc[t] = __builtin_amdgcn_wmma_f32_16x16x32_f16(
                             false, a, false, b, (short)0, acc[t], false, false);
            }
        }
    }

    const int mhi = (lane >> 4) << 3;
    #pragma unroll
    for (int t = 0; t < NT; ++t) {
        const int n = n0 + t * 16 + (lane & 15);
        if (n < CoutValid) {
            const float bv = bias ? bias[n] : 0.0f;
            #pragma unroll
            for (int r = 0; r < 8; ++r) {
                const int pp = pbase + r + mhi;
                if (pp < HW) Y[(size_t)pp * ldc + n] = acc[t][r] + bv;
            }
        }
    }
}

__global__ void zero_half_kernel(_Float16* __restrict__ p, int n)
{
    const int i = blockIdx.x * blockDim.x + threadIdx.x;
    if (i < n) p[i] = (_Float16)0.f;
}

// GroupNorm stats: one block per group (8 channels), Y is NHWC f32 [HW][256].
__global__ void gn_stats_kernel(const float* __restrict__ Y,
                                float* __restrict__ stats, int HW)
{
    __shared__ float s_sum[256], s_ss[256];
    const int g = blockIdx.x, tid = threadIdx.x;
    const int total = HW * 8;
    float sum = 0.f, ss = 0.f;
    for (int idx = tid; idx < total; idx += 256) {
        const int p = idx >> 3, c = (g << 3) + (idx & 7);
        const float v = Y[(size_t)p * 256 + c];
        sum += v; ss += v * v;
    }
    s_sum[tid] = sum; s_ss[tid] = ss;
    __syncthreads();
    for (int s = 128; s > 0; s >>= 1) {
        if (tid < s) { s_sum[tid] += s_sum[tid + s]; s_ss[tid] += s_ss[tid + s]; }
        __syncthreads();
    }
    if (tid == 0) {
        const float mean = s_sum[0] / (float)total;
        const float var  = s_ss[0] / (float)total - mean * mean;
        stats[g * 2]     = mean;
        stats[g * 2 + 1] = rsqrtf(var + 1e-5f);
    }
}

// Normalize + gamma/beta + ReLU + f16 convert (NHWC in/out).
__global__ void gn_apply_kernel(const float* __restrict__ Y,
                                const float* __restrict__ stats,
                                const float* __restrict__ gamma,
                                const float* __restrict__ beta,
                                _Float16* __restrict__ Xo, int HW)
{
    const int idx = blockIdx.x * blockDim.x + threadIdx.x;
    if (idx >= HW * 256) return;
    const int c = idx & 255, g = c >> 3;
    float v = (Y[idx] - stats[g * 2]) * stats[g * 2 + 1] * gamma[c] + beta[c];
    v = v > 0.f ? v : 0.f;
    Xo[idx] = (_Float16)v;
}

// f32 NCHW -> f16 NHWC
__global__ void nchw_to_nhwc_half(const float* __restrict__ F,
                                  _Float16* __restrict__ Xo, int HW)
{
    const int idx = blockIdx.x * blockDim.x + threadIdx.x;
    if (idx >= HW * 256) return;
    const int c = idx & 255, p = idx >> 8;
    Xo[idx] = (_Float16)F[(size_t)c * HW + p];
}

// Pack conv weights [O][256][3][3] f32 -> [k=tap*256+ci][Opad] f16 panel.
__global__ void pack_w_kernel(const float* __restrict__ Wsrc,
                              _Float16* __restrict__ Wp, int O, int Opad)
{
    const int idx = blockIdx.x * blockDim.x + threadIdx.x;
    if (idx >= 2304 * Opad) return;
    const int o = idx % Opad, k = idx / Opad;
    const int ci = k & 255, tap = k >> 8;
    _Float16 v = (_Float16)0.f;
    if (o < O) v = (_Float16)Wsrc[((size_t)o * 256 + ci) * 9 + tap];
    Wp[idx] = v;
}

// Pack pred(4ch)+iou(1ch) weights into one 16-wide padded panel.
__global__ void pack_predio_kernel(const float* __restrict__ pw,
                                   const float* __restrict__ iw,
                                   _Float16* __restrict__ Wp)
{
    const int idx = blockIdx.x * blockDim.x + threadIdx.x;
    if (idx >= 2304 * 16) return;
    const int o = idx & 15, k = idx >> 4;
    const int ci = k & 255, tap = k >> 8;
    float v = 0.f;
    if (o < 4)       v = pw[((size_t)o * 256 + ci) * 9 + tap];
    else if (o == 4) v = iw[(size_t)ci * 9 + tap];
    Wp[idx] = (_Float16)v;
}

// Box-head epilogue: out[p][80..83] = relu((pred+b)*scale)*stride,
//                    out[p][84]     = iou + b.
__global__ void pack_box_kernel(const float* __restrict__ S2,
                                const float* __restrict__ pred_b,
                                const float* __restrict__ iou_b,
                                const float* __restrict__ scales,
                                int level, float stride,
                                float* __restrict__ out, int HW)
{
    const int idx = blockIdx.x * blockDim.x + threadIdx.x;
    if (idx >= HW * 5) return;
    const int p = idx / 5, j = idx % 5;
    float* o = out + (size_t)p * 85;
    if (j < 4) {
        float v = (S2[(size_t)p * 16 + j] + pred_b[j]) * scales[level];
        v = v > 0.f ? v : 0.f;
        o[80 + j] = v * stride;
    } else {
        o[84] = S2[(size_t)p * 16 + 4] + iou_b[0];
    }
}

extern "C" void kernel_launch(void* const* d_in, const int* in_sizes, int n_in,
                              void* d_out, int out_size, void* d_ws, size_t ws_size,
                              hipStream_t stream)
{
    (void)in_sizes; (void)n_in; (void)out_size; (void)ws_size;
    const float* feats[5] = {(const float*)d_in[0], (const float*)d_in[1],
                             (const float*)d_in[2], (const float*)d_in[3],
                             (const float*)d_in[4]};
    const float* cls_w    = (const float*)d_in[5];
    const float* cls_b    = (const float*)d_in[6];
    const float* cls_gn_w = (const float*)d_in[7];
    const float* cls_gn_b = (const float*)d_in[8];
    const float* box_w    = (const float*)d_in[9];
    const float* box_b    = (const float*)d_in[10];
    const float* box_gn_w = (const float*)d_in[11];
    const float* box_gn_b = (const float*)d_in[12];
    const float* score_w  = (const float*)d_in[13];
    const float* score_b  = (const float*)d_in[14];
    const float* pred_w   = (const float*)d_in[15];
    const float* pred_b   = (const float*)d_in[16];
    const float* iou_w    = (const float*)d_in[17];
    const float* iou_b    = (const float*)d_in[18];
    const float* scales   = (const float*)d_in[19];
    float* out = (float*)d_out;

    // ---- workspace carve-up -------------------------------------------------
    char* ws = (char*)d_ws;
    size_t off = 0;
    auto take = [&](size_t b) -> char* {
        char* q = ws + off;
        off += (b + 255) & ~(size_t)255;
        return q;
    };
    _Float16* WpCls[4]; _Float16* WpBox[4];
    for (int i = 0; i < 4; ++i) WpCls[i] = (_Float16*)take((size_t)2304 * 256 * 2);
    for (int i = 0; i < 4; ++i) WpBox[i] = (_Float16*)take((size_t)2304 * 256 * 2);
    _Float16* WpScore  = (_Float16*)take((size_t)2304 * 80 * 2);
    _Float16* WpPredIo = (_Float16*)take((size_t)2304 * 16 * 2);
    _Float16* X0 = (_Float16*)take((size_t)6400 * 256 * 2);
    _Float16* XA = (_Float16*)take((size_t)6400 * 256 * 2);
    _Float16* XB = (_Float16*)take((size_t)6400 * 256 * 2);
    float*    Yb    = (float*)take((size_t)6400 * 256 * 4);
    float*    S2    = (float*)take((size_t)6400 * 16 * 4);
    float*    stats = (float*)take(256);
    _Float16* Zrow  = (_Float16*)take((size_t)512 * 2);

    // ---- zero row + weight packing -----------------------------------------
    zero_half_kernel<<<2, 256, 0, stream>>>(Zrow, 512);
    {
        const int blks = (2304 * 256) / 256;
        for (int i = 0; i < 4; ++i) {
            pack_w_kernel<<<blks, 256, 0, stream>>>(cls_w + (size_t)i * 256 * 256 * 9,
                                                    WpCls[i], 256, 256);
            pack_w_kernel<<<blks, 256, 0, stream>>>(box_w + (size_t)i * 256 * 256 * 9,
                                                    WpBox[i], 256, 256);
        }
        pack_w_kernel<<<(2304 * 80 + 255) / 256, 256, 0, stream>>>(score_w, WpScore, 80, 80);
        pack_predio_kernel<<<(2304 * 16) / 256, 256, 0, stream>>>(pred_w, iou_w, WpPredIo);
    }

    auto launch_tower_conv = [&](const _Float16* X, const _Float16* Wp,
                                 const float* bias, float* Y, int HW, int Wd) {
        const int mt = (HW + 15) / 16;
        const int waves = mt * 4;
        conv3x3_wmma_tdm<<<(waves + 3) / 4, 128, 0, stream>>>(
            X, Wp, Zrow, bias, Y, HW, Wd);
    };
    auto launch_head_conv = [&](const _Float16* X, const _Float16* Wp,
                                const float* bias, float* Y, int HW, int Wd,
                                int Cout, int ldc, int CoutValid) {
        const int mt = (HW + 15) / 16;
        if (Cout == 80) {
            conv3x3_wmma<5><<<(mt + 3) / 4, 128, 0, stream>>>(
                X, Wp, Zrow, bias, Y, HW, Wd, Cout, ldc, CoutValid);
        } else {  // Cout == 16
            conv3x3_wmma<1><<<(mt + 3) / 4, 128, 0, stream>>>(
                X, Wp, Zrow, bias, Y, HW, Wd, Cout, ldc, CoutValid);
        }
    };

    const int   dims[5]    = {80, 40, 20, 10, 5};
    const float fstride[5] = {8.f, 16.f, 32.f, 64.f, 128.f};
    int pixoff = 0;
    for (int l = 0; l < 5; ++l) {
        const int Wd = dims[l], HW = Wd * Wd;
        const int nel = HW * 256;
        nchw_to_nhwc_half<<<(nel + 255) / 256, 256, 0, stream>>>(feats[l], X0, HW);

        // cls tower -> score head (written directly into packed output, ldc=85)
        const _Float16* cur = X0;
        for (int i = 0; i < 4; ++i) {
            launch_tower_conv(cur, WpCls[i], cls_b + i * 256, Yb, HW, Wd);
            gn_stats_kernel<<<32, 256, 0, stream>>>(Yb, stats, HW);
            _Float16* nxt = (i & 1) ? XB : XA;
            gn_apply_kernel<<<(nel + 255) / 256, 256, 0, stream>>>(
                Yb, stats, cls_gn_w + i * 256, cls_gn_b + i * 256, nxt, HW);
            cur = nxt;
        }
        launch_head_conv(cur, WpScore, score_b, out + (size_t)pixoff * 85,
                         HW, Wd, 80, 85, 80);

        // box tower -> pred(4)+iou(1) head
        cur = X0;
        for (int i = 0; i < 4; ++i) {
            launch_tower_conv(cur, WpBox[i], box_b + i * 256, Yb, HW, Wd);
            gn_stats_kernel<<<32, 256, 0, stream>>>(Yb, stats, HW);
            _Float16* nxt = (i & 1) ? XB : XA;
            gn_apply_kernel<<<(nel + 255) / 256, 256, 0, stream>>>(
                Yb, stats, box_gn_w + i * 256, box_gn_b + i * 256, nxt, HW);
            cur = nxt;
        }
        launch_head_conv(cur, WpPredIo, nullptr, S2, HW, Wd, 16, 16, 16);
        pack_box_kernel<<<(HW * 5 + 255) / 256, 256, 0, stream>>>(
            S2, pred_b, iou_b, scales, l, fstride[l], out + (size_t)pixoff * 85, HW);

        pixoff += HW;
    }
}
